// MultiHeadAttention_5325759447452
// MI455X (gfx1250) — matrix operand order repaired and checked by reading, not compiled, not run on hardware
//
#include <hip/hip_runtime.h>
#include <hip/hip_bf16.h>

// ---------------------------------------------------------------------------
// MultiHeadAttention forward for MI455X (gfx1250, wave32, WMMA).
//   Kernel 1: proj  -- Q/K/V projection GEMMs (bf16 out, head-major),
//             16-B vector global loads, fp32->bf16 in registers.
//   Kernel 2: fused -- per 32-query-row block: K@Q^T (WMMA, S^T tiles) ->
//             softmax in LDS -> single NT write of attn -> P@V (WMMA) -> out.
// K/V tiles double-buffered via global_load_async_to_lds_b128 with
// s_wait_asynccnt 2 (overlap copy of tile t+1 with WMMAs on tile t).
// attn (537 MB) crosses HBM exactly once.
// ---------------------------------------------------------------------------

#define USE_ASYNC_STAGE 1

typedef __attribute__((ext_vector_type(16))) __bf16 v16bf;
typedef __attribute__((ext_vector_type(8)))  float  v8f;
typedef __attribute__((ext_vector_type(4)))  float  fx4;

#define DEV_INLINE __device__ __forceinline__

constexpr int B_   = 8;
constexpr int L_   = 1024;
constexpr int D_   = 1024;
constexpr int H_   = 16;
constexpr int DK_  = 64;
constexpr int TS   = 64;    // GEMM tile edge
constexpr int STR  = 72;    // padded LDS stride, bf16 elems (144 B, 16B-aligned)
constexpr int ROWS = 32;    // query rows per fused workgroup
constexpr int SSTR = 1028;  // fp32 score-strip stride; 1028%64==4 -> conflict-free
constexpr int NT_  = L_ / TS;  // 16 K/V tiles per (b,h)

// fp32 -> bf16, round-to-nearest-even
DEV_INLINE __bf16 f2bf(float x) {
  unsigned u = __float_as_uint(x);
  u += 0x7FFFu + ((u >> 16) & 1u);
  unsigned short s = (unsigned short)(u >> 16);
  __bf16 r;
  __builtin_memcpy(&r, &s, 2);
  return r;
}

DEV_INLINE v8f wmma_bf16(v16bf a, v16bf b, v8f c) {
  return __builtin_amdgcn_wmma_f32_16x16x32_bf16(false, a, false, b, (short)0, c,
                                                 false, false);
}

// A fragment (16x32 bf16): lane%16 = M row; lane/16 selects K halves
// {0-7,16-23} vs {8-15,24-31}. Two 16-byte LDS loads.
DEV_INLINE v16bf frag_a(const __bf16* t, int row, int kbase, int lane) {
  const int m = lane & 15, hh = lane >> 4;
  const __bf16* p = t + (row + m) * STR + kbase + 8 * hh;
  union { uint4 u[2]; v16bf v; } r;
  r.u[0] = *(const uint4*)(p);
  r.u[1] = *(const uint4*)(p + 16);
  return r.v;
}

// Same A fragment from an fp32 LDS strip (stride SSTR), bf16-converting.
DEV_INLINE v16bf frag_a_f32(const float* t, int row, int kbase, int lane) {
  const int m = lane & 15, hh = lane >> 4;
  const float* p = t + (size_t)(row + m) * SSTR + kbase + 8 * hh;
  fx4 a0 = *(const fx4*)(p);
  fx4 a1 = *(const fx4*)(p + 4);
  fx4 a2 = *(const fx4*)(p + 16);
  fx4 a3 = *(const fx4*)(p + 20);
  union { __bf16 e[16]; v16bf v; } r;
  r.e[0]  = f2bf(a0.x); r.e[1]  = f2bf(a0.y); r.e[2]  = f2bf(a0.z); r.e[3]  = f2bf(a0.w);
  r.e[4]  = f2bf(a1.x); r.e[5]  = f2bf(a1.y); r.e[6]  = f2bf(a1.z); r.e[7]  = f2bf(a1.w);
  r.e[8]  = f2bf(a2.x); r.e[9]  = f2bf(a2.y); r.e[10] = f2bf(a2.z); r.e[11] = f2bf(a2.w);
  r.e[12] = f2bf(a3.x); r.e[13] = f2bf(a3.y); r.e[14] = f2bf(a3.z); r.e[15] = f2bf(a3.w);
  return r.v;
}

// B fragment (32x16 bf16): lane%16 = N column; lane/16 selects K halves
// {0-7,16-23} vs {8-15,24-31}.
DEV_INLINE v16bf frag_b(const __bf16* t, int kbase, int col, int lane) {
  const int n = lane & 15, hh = lane >> 4;
  const __bf16* p = t + (kbase + 8 * hh) * STR + col + n;
  union { __bf16 e[16]; v16bf v; } r;
#pragma unroll
  for (int s = 0; s < 8; ++s) {
    r.e[s] = p[s * STR];
    r.e[s + 8] = p[(s + 16) * STR];
  }
  return r.v;
}

// Stage a 64x64 bf16 tile (row stride DK_ in global) into LDS (stride STR).
// 512 x 16-byte lane copies; async path writes LDS directly (no VGPR bounce),
// issuing exactly 2 ASYNCcnt ops per thread (enables wait<2> double buffering).
DEV_INLINE void stage_tile64(const __bf16* __restrict__ gsrc, __bf16* lds, int tid) {
#pragma unroll
  for (int it = 0; it < 2; ++it) {
    const int i = tid + it * 256;         // 0..511
    const int r = i >> 3, c8 = (i & 7) * 8;
    const __bf16* g = gsrc + ((size_t)r << 6) + c8;
    __bf16* l = lds + r * STR + c8;
#if USE_ASYNC_STAGE
    const unsigned loff = (unsigned)(size_t)l;   // generic VA low 32b == LDS addr
    asm volatile("global_load_async_to_lds_b128 %0, %1, off"
                 :: "v"(loff), "v"(g) : "memory");
#else
    *(uint4*)l = *(const uint4*)g;
#endif
  }
}

template <int N>
DEV_INLINE void stage_wait() {
#if USE_ASYNC_STAGE
  asm volatile("s_wait_asynccnt %0" :: "i"(N) : "memory");
#endif
}

// ---------------------------------------------------------------------------
// Kernel 1: fused Q/K/V projection.  X[b] (1024x1024) @ W (1024x1024),
// bf16 output in head-major layout [B][H][L][DK].
// grid = (D/64, L/64, 3*B); block = 256 (8 waves, 2 WMMA subtiles each).
// Staging: fx4 (16-B) global loads, packed 8-B bf16 LDS stores.
// ---------------------------------------------------------------------------
__global__ __launch_bounds__(256) void proj_kernel(
    const float* __restrict__ q, const float* __restrict__ k,
    const float* __restrict__ v, const float* __restrict__ wq,
    const float* __restrict__ wk, const float* __restrict__ wv,
    __bf16* __restrict__ qh, __bf16* __restrict__ kh, __bf16* __restrict__ vh) {
  __shared__ alignas(16) __bf16 Xs[TS * STR];
  __shared__ alignas(16) __bf16 Ws[TS * STR];
  const int n0 = blockIdx.x * TS;
  const int m0 = blockIdx.y * TS;
  const int z = blockIdx.z;
  const int mat = z >> 3;   // 0=q, 1=k, 2=v  (B_ == 8)
  const int b   = z & 7;
  const float* X = (mat == 0) ? q  : (mat == 1) ? k  : v;
  const float* W = (mat == 0) ? wq : (mat == 1) ? wk : wv;
  __bf16*    dst = (mat == 0) ? qh : (mat == 1) ? kh : vh;

  const int tid  = threadIdx.x;
  const int wave = tid >> 5, lane = tid & 31;
  const int mi  = wave >> 1;        // 0..3
  const int ni0 = (wave & 1) * 2;   // 0 or 2
  v8f acc0 = {}, acc1 = {};

  for (int k0 = 0; k0 < D_; k0 += TS) {
    // 64x64 fp32 tiles as fx4 vector loads; 4 bf16 packed per 8-B LDS store.
    for (int i = tid; i < TS * TS / 4; i += 256) {
      const int r = i >> 4, c4 = (i & 15) * 4;
      fx4 xv = *(const fx4*)&X[((size_t)b * L_ + (m0 + r)) * D_ + (k0 + c4)];
      fx4 wv4 = *(const fx4*)&W[(size_t)(k0 + r) * D_ + (n0 + c4)];
      union { __bf16 e[4]; unsigned long long u; } px, pw;
      px.e[0] = f2bf(xv.x);  px.e[1] = f2bf(xv.y);
      px.e[2] = f2bf(xv.z);  px.e[3] = f2bf(xv.w);
      pw.e[0] = f2bf(wv4.x); pw.e[1] = f2bf(wv4.y);
      pw.e[2] = f2bf(wv4.z); pw.e[3] = f2bf(wv4.w);
      *(unsigned long long*)&Xs[r * STR + c4] = px.u;
      *(unsigned long long*)&Ws[r * STR + c4] = pw.u;
    }
    __syncthreads();
#pragma unroll
    for (int kk = 0; kk < TS; kk += 32) {
      v16bf a  = frag_a(Xs, mi * 16, kk, lane);
      v16bf b0 = frag_b(Ws, kk, ni0 * 16, lane);
      v16bf b1 = frag_b(Ws, kk, ni0 * 16 + 16, lane);
      acc0 = wmma_bf16(a, b0, acc0);
      acc1 = wmma_bf16(a, b1, acc1);
    }
    __syncthreads();
  }

  const int M    = m0 + mi * 16 + 8 * (lane >> 4);
  const int head = n0 >> 6;
  const int dk_a = (ni0 * 16) + (lane & 15);
  const int dk_b = dk_a + 16;
  __bf16* obase = dst + ((size_t)(b * H_ + head) * L_) * DK_;
#pragma unroll
  for (int g = 0; g < 8; ++g) {
    obase[(size_t)(M + g) * DK_ + dk_a] = f2bf(acc0[g]);
    obase[(size_t)(M + g) * DK_ + dk_b] = f2bf(acc1[g]);
  }
}

// ---------------------------------------------------------------------------
// Kernel 2: fused attention for one (b, h, 32-row query block).
// grid = (L/32, B*H); block = 256 (8 waves).
// Scores computed transposed (A = K tile, B = Q^T) so K staging is a straight
// row copy -> async-to-LDS; D = S^T lands contiguously along n per lane
// -> two b128 LDS stores per accumulator.  K/V tiles ping-pong buffered:
// issue tile t+1, s_wait_asynccnt 2 (tile t landed), barrier, compute.
// ---------------------------------------------------------------------------
__global__ __launch_bounds__(256) void attn_fused_kernel(
    const __bf16* __restrict__ qh, const __bf16* __restrict__ kh,
    const __bf16* __restrict__ vh, float* __restrict__ attn,
    float* __restrict__ out) {
  __shared__ alignas(16) float  Sbuf[ROWS * SSTR];   // fp32 score strip [m][n]
  __shared__ alignas(16) __bf16 QTs[TS * STR];       // Q^T block [dk][m]
  __shared__ alignas(16) __bf16 KVs[2][TS * STR];    // ping-pong K/V tiles
  __shared__ float red[256];
  __shared__ float rowmax[ROWS];
  __shared__ float rowinv[ROWS];

  const int bh = blockIdx.y;
  const int b = bh >> 4, head = bh & 15;
  const int m0 = blockIdx.x * ROWS;
  const __bf16* Q = qh + (size_t)bh * L_ * DK_;
  const __bf16* K = kh + (size_t)bh * L_ * DK_;
  const __bf16* V = vh + (size_t)bh * L_ * DK_;

  const int tid = threadIdx.x, wave = tid >> 5, lane = tid & 31;

  // ---- Phase 1: stage Q^T (dk x 32) once.  One b128 global load per thread,
  // 8 b16 scatter stores.  Synced by the first loop barrier.
  {
    const int r = tid >> 3, cb = (tid & 7) * 8;   // 32 rows x 8 chunks == 256
    union { uint4 u; __bf16 e[8]; } qv;
    qv.u = *(const uint4*)&Q[(size_t)(m0 + r) * DK_ + cb];
#pragma unroll
    for (int j = 0; j < 8; ++j) QTs[(cb + j) * STR + r] = qv.e[j];
  }

  // ---- Phase 2: S^T tiles = K_tile @ Q^T, scaled, into Sbuf[m][n] (fp32).
  {
    const int niA = wave >> 1;   // 0..3 : 16-key subtile (M dim of D)
    const int miB = wave & 1;    // 0..1 : 16-query subtile (N dim of D)
    stage_tile64(K, KVs[0], tid);
    for (int t = 0; t < NT_; ++t) {
      const int n0 = t * TS;
      const __bf16* cur = KVs[t & 1];
      if (t + 1 < NT_) {
        stage_tile64(K + (size_t)(n0 + TS) * DK_, KVs[(t + 1) & 1], tid);
        stage_wait<2>();   // tile t complete; tile t+1 still in flight
      } else {
        stage_wait<0>();
      }
      __syncthreads();
      v8f acc = {};
#pragma unroll
      for (int kk = 0; kk < DK_; kk += 32) {
        v16bf a  = frag_a(cur, niA * 16, kk, lane);   // K rows (n) x dk
        v16bf bb = frag_b(QTs, kk, miB * 16, lane);   // dk x m
        acc = wmma_bf16(a, bb, acc);
      }
      // D element (g, lane): n = n0 + niA*16 + 8*(lane>>4) + g ; m = miB*16 + lane%16
      const int srow = miB * 16 + (lane & 15);
      const int scol = n0 + niA * 16 + 8 * (lane >> 4);
      float* sp = Sbuf + (size_t)srow * SSTR + scol;
      fx4 lo, hi;
      lo.x = acc[0] * 0.125f; lo.y = acc[1] * 0.125f;
      lo.z = acc[2] * 0.125f; lo.w = acc[3] * 0.125f;
      hi.x = acc[4] * 0.125f; hi.y = acc[5] * 0.125f;
      hi.z = acc[6] * 0.125f; hi.w = acc[7] * 0.125f;
      *(fx4*)sp = lo;
      *((fx4*)sp + 1) = hi;
      __syncthreads();
    }
  }

  // ---- Phase 3: row softmax in LDS.  8 threads per row, 128 cols each.
  {
    const int r = tid >> 3, j = tid & 7;
    float* prow = Sbuf + (size_t)r * SSTR + j * 128;
    float mx = -3.4e38f;
#pragma unroll 8
    for (int c4 = 0; c4 < 32; ++c4) {
      fx4 x = *(const fx4*)(prow + c4 * 4);
      mx = fmaxf(mx, fmaxf(fmaxf(x.x, x.y), fmaxf(x.z, x.w)));
    }
    red[tid] = mx;
    __syncthreads();
    if (j == 0) {
      float m2 = red[tid];
#pragma unroll
      for (int t = 1; t < 8; ++t) m2 = fmaxf(m2, red[tid + t]);
      rowmax[r] = m2;
    }
    __syncthreads();
    const float mrow = rowmax[r];
    float sum = 0.f;
#pragma unroll 8
    for (int c4 = 0; c4 < 32; ++c4) {
      fx4 x = *(const fx4*)(prow + c4 * 4);
      x.x = __expf(x.x - mrow); x.y = __expf(x.y - mrow);
      x.z = __expf(x.z - mrow); x.w = __expf(x.w - mrow);
      *(fx4*)(prow + c4 * 4) = x;
      sum += x.x + x.y + x.z + x.w;
    }
    red[tid] = sum;
    __syncthreads();
    if (j == 0) {
      float s2 = red[tid];
#pragma unroll
      for (int t = 1; t < 8; ++t) s2 += red[tid + t];
      rowinv[r] = 1.0f / s2;
    }
    __syncthreads();
  }

  // ---- Phase 3b: normalize in LDS + single non-temporal write of attn.
  {
    float* Abase = attn + (size_t)bh * L_ * L_ + (size_t)m0 * L_;
    for (int i = tid; i < ROWS * (L_ / 4); i += 256) {
      const int rr = i >> 8;        // L_/4 == 256
      const int c4 = i & 255;
      fx4 x = *(const fx4*)(Sbuf + (size_t)rr * SSTR + c4 * 4);
      const float s = rowinv[rr];
      x.x *= s; x.y *= s; x.z *= s; x.w *= s;
      *(fx4*)(Sbuf + (size_t)rr * SSTR + c4 * 4) = x;
      __builtin_nontemporal_store(x, (fx4*)(Abase + (size_t)rr * L_) + c4);
    }
  }

  // ---- Phase 4: out = P @ V, K-loop over Lk, P read from fp32 LDS strip.
  const int mi4 = wave >> 2;   // 0..1
  const int ni4 = wave & 3;    // 0..3
  v8f accO = {};
  stage_tile64(V, KVs[0], tid);
  for (int t = 0; t < NT_; ++t) {
    const int k0 = t * TS;
    const __bf16* cur = KVs[t & 1];
    if (t + 1 < NT_) {
      stage_tile64(V + (size_t)(k0 + TS) * DK_, KVs[(t + 1) & 1], tid);
      stage_wait<2>();
    } else {
      stage_wait<0>();
    }
    __syncthreads();   // tile t landed; P normalized (first iteration)
#pragma unroll
    for (int kk = 0; kk < TS; kk += 32) {
      v16bf a  = frag_a_f32(Sbuf, mi4 * 16, k0 + kk, lane);
      v16bf bb = frag_b(cur, kk, ni4 * 16, lane);
      accO = wmma_bf16(a, bb, accO);
    }
    __syncthreads();   // tile t consumed before buffer reuse at t+2
  }

  const int Mo = m0 + mi4 * 16 + 8 * (lane >> 4);
  const int dk = ni4 * 16 + (lane & 15);
#pragma unroll
  for (int g = 0; g < 8; ++g)
    __builtin_nontemporal_store(
        accO[g], &out[((size_t)b * L_ + (Mo + g)) * D_ + head * DK_ + dk]);
}

// ---------------------------------------------------------------------------
extern "C" void kernel_launch(void* const* d_in, const int* in_sizes, int n_in,
                              void* d_out, int out_size, void* d_ws, size_t ws_size,
                              hipStream_t stream) {
  (void)in_sizes; (void)n_in; (void)out_size; (void)ws_size;
  const float* q  = (const float*)d_in[0];
  const float* k  = (const float*)d_in[1];
  const float* v  = (const float*)d_in[2];
  const float* wq = (const float*)d_in[3];
  const float* wk = (const float*)d_in[4];
  const float* wv = (const float*)d_in[5];

  float* out  = (float*)d_out;                  // [B, L, D] fp32
  float* attn = out + (size_t)B_ * L_ * D_;     // [B, H, L, L] fp32

  // Workspace: bf16 head-major projections, 3 x 16 MB.
  __bf16* qh = (__bf16*)d_ws;
  __bf16* kh = qh + (size_t)B_ * H_ * L_ * DK_;
  __bf16* vh = kh + (size_t)B_ * H_ * L_ * DK_;

  proj_kernel<<<dim3(D_ / 64, L_ / 64, 3 * B_), 256, 0, stream>>>(
      q, k, v, wq, wk, wv, qh, kh, vh);
  attn_fused_kernel<<<dim3(L_ / ROWS, B_ * H_), 256, 0, stream>>>(
      qh, kh, vh, attn, out);
}
